// HierarchicalGNNBlock_7559142441636
// MI455X (gfx1250) — matrix operand order repaired and checked by reading, not compiled, not run on hardware
//
#include <hip/hip_runtime.h>
#include <hip/hip_bf16.h>
#include <math.h>

#define LATENT   128
#define HID      512
#define NN_NODES 100000
#define NN_EDGES 500000
#define NN_SUPER 2000
#define NN_BIP   400000
#define NN_SEDG  64000

typedef __attribute__((ext_vector_type(16))) __bf16         v16bf;
typedef __attribute__((ext_vector_type(8)))  unsigned int   v8u;
typedef __attribute__((ext_vector_type(8)))  float          v8f;

// LDS row stride for the bf16 A-operand buffer: 520 ushorts = 260 dwords.
// 260 mod 64 = 4 -> the 16 A-frag lanes land on disjoint 4-bank quads for ds_load_b128.
#define XLD 520

// ---------------- scalar helpers ----------------
__device__ __forceinline__ unsigned short f2bf(float f) {
  unsigned u = __builtin_bit_cast(unsigned, f);
  u += 0x7fffu + ((u >> 16) & 1u);          // round-to-nearest-even
  return (unsigned short)(u >> 16);
}
__device__ __forceinline__ float bf2f(unsigned short h) {
  unsigned u = ((unsigned)h) << 16;
  return __builtin_bit_cast(float, u);
}
__device__ __forceinline__ float gelu_f(float x) {
  const float c0 = 0.7978845608028654f;     // sqrt(2/pi), tanh-approx gelu (jax default)
  float x3 = x * x * x;
  return 0.5f * x * (1.0f + tanhf(c0 * (x + 0.044715f * x3)));
}

// ---------------- WMMA fragment loaders (ISA 7.12.2 layouts, wave32) ----------------
// A (16xK bf16), rows row0..row0+15 of Xa: lane L<16 -> row row0+L, VGPR v<4: K=2v,2v+1;
// v>=4: K=16+2(v-4); lanes>=16 add +8 to K.
__device__ __forceinline__ v16bf load_a_frag(const unsigned short* Xa, int kbase, int lane, int row0) {
  int row   = row0 + (lane & 15);
  int khalf = (lane >> 4) << 3;
  const unsigned* q = (const unsigned*)(Xa + row * XLD + kbase + khalf);
  v8u f;
#pragma unroll
  for (int v = 0; v < 4; ++v) f[v] = q[v];        // K  0..7  (+khalf)
#pragma unroll
  for (int v = 0; v < 4; ++v) f[4 + v] = q[8 + v];// K 16..23 (+khalf)
  return __builtin_bit_cast(v16bf, f);
}
// B (Kx16 bf16): lane L -> col N = ntile*16 + (L&15); elems 0..7: K 0..7, 8..15: K 16..23; lanes>=16 add +8 to K.
// Wt is transposed weights Wt[n*KD + k] (bf16 bits, L2-resident).
__device__ __forceinline__ v16bf load_b_frag(const unsigned short* Wt, int KD, int ntile, int kbase, int lane) {
  int n     = (ntile << 4) + (lane & 15);
  int khalf = (lane >> 4) << 3;
  const unsigned* q = (const unsigned*)(Wt + (size_t)n * KD + kbase + khalf);
  v8u f;
#pragma unroll
  for (int v = 0; v < 4; ++v) f[v] = q[v];
#pragma unroll
  for (int v = 0; v < 4; ++v) f[4 + v] = q[8 + v];
  return __builtin_bit_cast(v16bf, f);
}

// Cross-lane row reduction helper: reduce (s,s2) over the 16 lanes of each EXEC half
// (lanes 0-15 / 16-31 hold different rows per the C/D layout, so masks 1..8 only).
__device__ __forceinline__ void half_reduce2(float& s, float& s2) {
#pragma unroll
  for (int m = 1; m < 16; m <<= 1) {
    s  += __shfl_xor(s,  m, 32);
    s2 += __shfl_xor(s2, m, 32);
  }
}

// ---------------- hidden layer: Xa(32xKD) @ Wt^T(KDx512) -> LN -> act -> Xa ----------------
// Two 16-row M-tiles per wave reuse each B fragment (halves L2 weight traffic per WMMA).
// LN stats come straight from accumulator registers (no f32 LDS round trip).
template<int KD, int HACT>
__device__ __forceinline__ void hidden_layer(unsigned short* Xa,
                                             const unsigned short* __restrict__ Wt,
                                             const float* __restrict__ bias,
                                             const float* __restrict__ g,
                                             const float* __restrict__ be,
                                             float red[2][32][8],
                                             float* rowm, float* rowr,
                                             int tid, int lane, int wave) {
  v8f acc[2][4];
#pragma unroll
  for (int h = 0; h < 2; ++h)
#pragma unroll
    for (int t = 0; t < 4; ++t) acc[h][t] = v8f{0.f,0.f,0.f,0.f,0.f,0.f,0.f,0.f};

  for (int kc = 0; kc < KD; kc += 32) {
    v16bf a0 = load_a_frag(Xa, kc, lane, 0);
    v16bf a1 = load_a_frag(Xa, kc, lane, 16);
#pragma unroll
    for (int t = 0; t < 4; ++t) {
      v16bf b = load_b_frag(Wt, KD, wave * 4 + t, kc, lane);
      acc[0][t] = __builtin_amdgcn_wmma_f32_16x16x32_bf16(false, a0, false, b, (short)0, acc[0][t], false, false);
      acc[1][t] = __builtin_amdgcn_wmma_f32_16x16x32_bf16(false, a1, false, b, (short)0, acc[1][t], false, false);
    }
  }
  // bias (LN statistics are over x@W + b)
#pragma unroll
  for (int t = 0; t < 4; ++t) {
    float bv = bias[(wave * 4 + t) * 16 + (lane & 15)];
#pragma unroll
    for (int h = 0; h < 2; ++h)
#pragma unroll
      for (int r = 0; r < 8; ++r) acc[h][t][r] += bv;
  }
  __syncthreads();   // all Xa reads of this layer done before any Xa rewrite below

  // per-row mean/var partials: lane holds rows mbase..mbase+7 (x2 halves), 4 cols each
  const int mbase = (lane >> 4) << 3;
#pragma unroll
  for (int h = 0; h < 2; ++h)
#pragma unroll
    for (int r = 0; r < 8; ++r) {
      float s = 0.f, s2 = 0.f;
#pragma unroll
      for (int t = 0; t < 4; ++t) { float x = acc[h][t][r]; s += x; s2 += x * x; }
      half_reduce2(s, s2);
      if ((lane & 15) == 0) {
        int row = h * 16 + mbase + r;
        red[0][row][wave] = s;
        red[1][row][wave] = s2;
      }
    }
  __syncthreads();
  if (tid < 32) {
    float a = 0.f, b = 0.f;
#pragma unroll
    for (int w = 0; w < 8; ++w) { a += red[0][tid][w]; b += red[1][tid][w]; }
    float m = a / (float)HID, v = b / (float)HID - m * m;
    rowm[tid] = m; rowr[tid] = rsqrtf(v + 1e-5f);
  }
  __syncthreads();

  // normalize + activate from registers, write bf16 A operand for next layer
#pragma unroll
  for (int t = 0; t < 4; ++t) {
    int nc = (wave * 4 + t) * 16 + (lane & 15);
    float gv = g[nc], bev = be[nc];
#pragma unroll
    for (int h = 0; h < 2; ++h)
#pragma unroll
      for (int r = 0; r < 8; ++r) {
        int row = h * 16 + mbase + r;
        float y = (acc[h][t][r] - rowm[row]) * rowr[row] * gv + bev;
        y = (HACT == 1) ? gelu_f(y) : tanhf(y);
        Xa[row * XLD + nc] = f2bf(y);
      }
  }
  __syncthreads();
}

// ---------------- fused MLP kernel: 32 rows per block, 8 waves ----------------
// x = concat(gather(s0,i0), gather(s1,i1)[, gather(s2,i2)])  (each 128 wide)
// h = hact(LN(x@W0+b0)); h = hact(LN(h@W1+b1));
// OUT==128: y = oact(LN(h@W2+b2)) + residual -> out
// OUT==1  : out = exp(h . w2f + b2 + extra)   (fused att = exp(logits))
template<int IN_DIM, int OUT_DIM, int HACT, int OACT>
__global__ __launch_bounds__(256)
void mlp_kernel(int n_rows,
                const float* __restrict__ s0, const int* __restrict__ i0,
                const float* __restrict__ s1, const int* __restrict__ i1,
                const float* __restrict__ s2, const int* __restrict__ i2,
                const unsigned short* __restrict__ Wt0,
                const unsigned short* __restrict__ Wt1,
                const unsigned short* __restrict__ Wt2,
                const float* __restrict__ w2f,
                const float* __restrict__ b0, const float* __restrict__ b1, const float* __restrict__ b2,
                const float* __restrict__ g0, const float* __restrict__ be0,
                const float* __restrict__ g1, const float* __restrict__ be1,
                const float* __restrict__ g2, const float* __restrict__ be2,
                const float* __restrict__ residual,
                const float* __restrict__ extra,
                float* __restrict__ out) {
  __shared__ unsigned short Xa[32 * XLD];   // 33280 B bf16 activations (A operand)
  __shared__ float red[2][32][8];           // 2 KB cross-wave LN partials
  __shared__ float rowm[32], rowr[32];

  const int tid  = threadIdx.x;
  const int lane = tid & 31;
  const int wave = tid >> 5;
  const int base = blockIdx.x * 32;

  __builtin_prefetch(Wt0, 0, 3);            // weights are L2-hot

  // ---- stage gathered+concatenated input as bf16 into LDS ----
  for (int idx = tid; idx < 32 * IN_DIM; idx += 256) {
    int r = idx / IN_DIM, c = idx - r * IN_DIM;
    int grow = base + r; if (grow >= n_rows) grow = n_rows - 1;
    int part = c >> 7, f = c & 127;
    const float* s = (part == 0) ? s0 : ((part == 1) ? s1 : s2);
    const int* ix  = (part == 0) ? i0 : ((part == 1) ? i1 : i2);
    int g = ix ? ix[grow] : grow;
    Xa[r * XLD + c] = f2bf(s[(size_t)g * LATENT + f]);
  }
  __syncthreads();

  hidden_layer<IN_DIM, HACT>(Xa, Wt0, b0, g0, be0, red, rowm, rowr, tid, lane, wave);
  hidden_layer<HID,    HACT>(Xa, Wt1, b1, g1, be1, red, rowm, rowr, tid, lane, wave);

  if constexpr (OUT_DIM == 128) {
    // ---- layer 3: 512 -> 128, wave w owns N-tile w, both row halves ----
    v8f acc[2];
    acc[0] = v8f{0.f,0.f,0.f,0.f,0.f,0.f,0.f,0.f};
    acc[1] = acc[0];
    for (int kc = 0; kc < HID; kc += 32) {
      v16bf a0 = load_a_frag(Xa, kc, lane, 0);
      v16bf a1 = load_a_frag(Xa, kc, lane, 16);
      v16bf b  = load_b_frag(Wt2, HID, wave, kc, lane);
      acc[0] = __builtin_amdgcn_wmma_f32_16x16x32_bf16(false, a0, false, b, (short)0, acc[0], false, false);
      acc[1] = __builtin_amdgcn_wmma_f32_16x16x32_bf16(false, a1, false, b, (short)0, acc[1], false, false);
    }
    const int mbase = (lane >> 4) << 3;
    const int nc = wave * 16 + (lane & 15);
    float bv = b2[nc];
#pragma unroll
    for (int h = 0; h < 2; ++h)
#pragma unroll
      for (int r = 0; r < 8; ++r) acc[h][r] += bv;

#pragma unroll
    for (int h = 0; h < 2; ++h)
#pragma unroll
      for (int r = 0; r < 8; ++r) {
        float s = acc[h][r], s2 = s * s;
        half_reduce2(s, s2);
        if ((lane & 15) == 0) {
          int row = h * 16 + mbase + r;
          red[0][row][wave] = s;
          red[1][row][wave] = s2;
        }
      }
    __syncthreads();
    if (tid < 32) {
      float a = 0.f, b = 0.f;
#pragma unroll
      for (int w = 0; w < 8; ++w) { a += red[0][tid][w]; b += red[1][tid][w]; }
      float m = a / 128.f, v = b / 128.f - m * m;
      rowm[tid] = m; rowr[tid] = rsqrtf(v + 1e-5f);
    }
    __syncthreads();
    float gv = g2[nc], bev = be2[nc];
#pragma unroll
    for (int h = 0; h < 2; ++h)
#pragma unroll
      for (int r = 0; r < 8; ++r) {
        int row  = h * 16 + mbase + r;
        int grow = base + row;
        if (grow < n_rows) {
          float y = (acc[h][r] - rowm[row]) * rowr[row] * gv + bev;
          y = (OACT == 1) ? gelu_f(y) : tanhf(y);
          if (residual) y += residual[(size_t)grow * LATENT + nc];
          out[(size_t)grow * LATENT + nc] = y;
        }
      }
  } else {
    // ---- layer 3: 512 -> 1 dot product, fused exp(logit + extra) ----
    int row = tid >> 3, seg = tid & 7;     // 8 threads per row, 64 elems each
    float s = 0.f;
#pragma unroll 8
    for (int c = seg * 64; c < seg * 64 + 64; ++c)
      s += bf2f(Xa[row * XLD + c]) * w2f[c];
    red[0][row][seg] = s;
    __syncthreads();
    if (tid < 32) {
      int grow = base + tid;
      if (grow < n_rows) {
        float a = 0.f;
#pragma unroll
        for (int k = 0; k < 8; ++k) a += red[0][tid][k];
        a += b2[0];
        if (extra) a += extra[grow];
        out[grow] = expf(a);
      }
    }
  }
}

// ---------------- small utility kernels ----------------
__global__ void cvt_wt(const float* __restrict__ W, unsigned short* __restrict__ Wt,
                       int in_dim, int out_dim) {
  int i = blockIdx.x * 256 + threadIdx.x;
  if (i >= in_dim * out_dim) return;
  int k = i / out_dim, n = i - k * out_dim;    // W is (in,out) row-major
  Wt[(size_t)n * in_dim + k] = f2bf(W[i]);     // store transposed bf16 (B operand)
}

__global__ void zero_f(float* p, size_t n) {
  size_t i = (size_t)blockIdx.x * 256 + threadIdx.x;
  if (i < n) p[i] = 0.f;
}

__global__ void seg_sum_1d(const float* __restrict__ att, const int* __restrict__ seg,
                           float* __restrict__ denom, int n) {
  int i = blockIdx.x * 256 + threadIdx.x;
  if (i < n) atomicAdd(&denom[seg[i]], att[i]);
}

__global__ void att_normalize(float* att, const float* __restrict__ denom,
                              const int* __restrict__ seg, int n) {
  int i = blockIdx.x * 256 + threadIdx.x;
  if (i < n) att[i] = att[i] / (1e-12f + denom[seg[i]]);
}

// dst[di[e]] += w[e] * src[si?si[e]:e]   (per-feature atomic scatter, f = lane%128)
__global__ void scatter_weighted(const float* __restrict__ w, const float* __restrict__ src,
                                 const int* __restrict__ si, const int* __restrict__ di,
                                 float* __restrict__ dst, int n_edges) {
  unsigned gid = blockIdx.x * 256 + threadIdx.x;
  int e = gid >> 7, f = gid & 127;
  if (e < n_edges) {
    int s = si ? si[e] : e;
    float wv = w ? w[e] : 1.f;
    atomicAdd(&dst[(size_t)di[e] * LATENT + f], wv * src[(size_t)s * LATENT + f]);
  }
}

// ---------------- host side ----------------
struct MlpP { const float *W0,*W1,*W2,*b0,*b1,*b2,*be0,*be1,*be2,*g0,*g1,*g2; };

extern "C" void kernel_launch(void* const* d_in, const int* in_sizes, int n_in,
                              void* d_out, int out_size, void* d_ws, size_t ws_size,
                              hipStream_t stream) {
  const float* nodes      = (const float*)d_in[0];
  const float* edges      = (const float*)d_in[1];
  const float* supernodes = (const float*)d_in[2];
  const float* superedges = (const float*)d_in[3];
  const int*   graph      = (const int*)d_in[4];      // [2][N_EDGES]
  const int*   bip_src    = (const int*)d_in[5];
  const int*   bip_dst    = (const int*)d_in[6];
  const int*   sgraph     = (const int*)d_in[7];      // [2][N_SEDGES]
  const float* bip_logits = (const float*)d_in[8];
  const float* sga        = (const float*)d_in[9];
  const int* g0i = graph, *g1i = graph + NN_EDGES;
  const int* sg0 = sgraph, *sg1 = sgraph + NN_SEDG;

  // params pytree-flattened: outer keys sorted (edge,node,score,superedge,supernode),
  // inner keys sorted (W[3], b[3], beta[n], g[n])
  auto F = [&](int i) { return (const float*)d_in[i]; };
  auto mlp_at = [&](int B) {
    MlpP p{F(B),F(B+1),F(B+2),F(B+3),F(B+4),F(B+5),F(B+6),F(B+7),F(B+8),F(B+9),F(B+10),F(B+11)};
    return p;
  };
  MlpP Pe  = mlp_at(10);   // edge
  MlpP Pn  = mlp_at(22);   // node
  // score: W 34..36, b 37..39, beta 40..41, g 42..43
  MlpP Ps{F(34),F(35),F(36),F(37),F(38),F(39),F(40),F(41),nullptr,F(42),F(43),nullptr};
  MlpP Pse = mlp_at(44);   // superedge
  MlpP Psn = mlp_at(56);   // supernode

  // ---- workspace carve ----
  char* ws = (char*)d_ws;
  size_t off = 0;
  auto carve = [&](size_t bytes) -> void* {
    off = (off + 255) & ~(size_t)255;
    void* p = ws + off; off += bytes; return p;
  };
  auto wt = [&](size_t n) { return (unsigned short*)carve(n * 2); };
  unsigned short *eW0=wt(384*512), *eW1=wt(512*512), *eW2=wt(512*128);
  unsigned short *nW0=wt(384*512), *nW1=wt(512*512), *nW2=wt(512*128);
  unsigned short *snW0=wt(384*512),*snW1=wt(512*512),*snW2=wt(512*128);
  unsigned short *seW0=wt(384*512),*seW1=wt(512*512),*seW2=wt(512*128);
  unsigned short *sW0=wt(256*512), *sW1=wt(512*512);
  float* att   = (float*)carve((size_t)NN_BIP * 4);
  float* denom = (float*)carve((size_t)NN_NODES * 4);
  float* nmsg  = (float*)carve((size_t)NN_SUPER * LATENT * 4);
  float* amsg  = (float*)carve((size_t)NN_SUPER * LATENT * 4);
  float* snmsg = (float*)carve((size_t)NN_NODES * LATENT * 4);
  float* emsg  = (float*)carve((size_t)NN_NODES * LATENT * 4);
  size_t zero_count = (size_t)(((char*)emsg + (size_t)NN_NODES*LATENT*4) - (char*)denom) / 4;

  // ---- d_out partitions: (nodes, edges, supernodes, superedges) ----
  float* out_nodes  = (float*)d_out;
  float* out_edges  = out_nodes + (size_t)NN_NODES * LATENT;
  float* out_super  = out_edges + (size_t)NN_EDGES * LATENT;
  float* out_sedges = out_super + (size_t)NN_SUPER * LATENT;

  // ---- 1) convert + transpose all GEMM weights to bf16 (L2-resident) ----
  struct Job { const float* W; unsigned short* Wt; int in, out; };
  Job jobs[14] = {
    {Pe.W0,eW0,384,512},{Pe.W1,eW1,512,512},{Pe.W2,eW2,512,128},
    {Pn.W0,nW0,384,512},{Pn.W1,nW1,512,512},{Pn.W2,nW2,512,128},
    {Psn.W0,snW0,384,512},{Psn.W1,snW1,512,512},{Psn.W2,snW2,512,128},
    {Pse.W0,seW0,384,512},{Pse.W1,seW1,512,512},{Pse.W2,seW2,512,128},
    {Ps.W0,sW0,256,512},{Ps.W1,sW1,512,512},
  };
  for (int j = 0; j < 14; ++j) {
    int n = jobs[j].in * jobs[j].out;
    cvt_wt<<<(n + 255) / 256, 256, 0, stream>>>(jobs[j].W, jobs[j].Wt, jobs[j].in, jobs[j].out);
  }

  // ---- 2) zero accumulators ----
  zero_f<<<(unsigned)((zero_count + 255) / 256), 256, 0, stream>>>(denom, zero_count);

  // ---- 3) score MLP: att = exp(bip_logits + mlp(concat(nodes[src], super[dst]))) ----
  mlp_kernel<256, 1, 0, 0><<<(NN_BIP + 31) / 32, 256, 0, stream>>>(
      NN_BIP, nodes, bip_src, supernodes, bip_dst, nullptr, nullptr,
      sW0, sW1, nullptr, Ps.W2,
      Ps.b0, Ps.b1, Ps.b2, Ps.g0, Ps.be0, Ps.g1, Ps.be1, nullptr, nullptr,
      nullptr, bip_logits, att);

  // ---- 4) softmax denominator over bipartite_src, then normalize ----
  seg_sum_1d<<<(NN_BIP + 255) / 256, 256, 0, stream>>>(att, bip_src, denom, NN_BIP);
  att_normalize<<<(NN_BIP + 255) / 256, 256, 0, stream>>>(att, denom, bip_src, NN_BIP);

  // ---- 5) node_messages and attention_messages scatters ----
  scatter_weighted<<<(unsigned)(((size_t)NN_BIP * LATENT + 255) / 256), 256, 0, stream>>>(
      att, nodes, bip_src, bip_dst, nmsg, NN_BIP);
  scatter_weighted<<<(unsigned)(((size_t)NN_SEDG * LATENT + 255) / 256), 256, 0, stream>>>(
      sga, superedges, sg0, sg1, amsg, NN_SEDG);

  // ---- 6) supernode MLP (+residual) -> out_super ----
  mlp_kernel<384, 128, 1, 1><<<(NN_SUPER + 31) / 32, 256, 0, stream>>>(
      NN_SUPER, supernodes, nullptr, amsg, nullptr, nmsg, nullptr,
      snW0, snW1, snW2, nullptr,
      Psn.b0, Psn.b1, Psn.b2, Psn.g0, Psn.be0, Psn.g1, Psn.be1, Psn.g2, Psn.be2,
      supernodes, nullptr, out_super);

  // ---- 7) supernode_messages and edge_messages scatters ----
  scatter_weighted<<<(unsigned)(((size_t)NN_BIP * LATENT + 255) / 256), 256, 0, stream>>>(
      att, out_super, bip_dst, bip_src, snmsg, NN_BIP);
  scatter_weighted<<<(unsigned)(((size_t)NN_EDGES * LATENT + 255) / 256), 256, 0, stream>>>(
      nullptr, edges, nullptr, g1i, emsg, NN_EDGES);

  // ---- 8) node MLP (+residual) -> out_nodes ----
  mlp_kernel<384, 128, 1, 1><<<(NN_NODES + 31) / 32, 256, 0, stream>>>(
      NN_NODES, nodes, nullptr, emsg, nullptr, snmsg, nullptr,
      nW0, nW1, nW2, nullptr,
      Pn.b0, Pn.b1, Pn.b2, Pn.g0, Pn.be0, Pn.g1, Pn.be1, Pn.g2, Pn.be2,
      nodes, nullptr, out_nodes);

  // ---- 9) superedge MLP (uses NEW supernodes, +residual, tanh out) -> out_sedges ----
  mlp_kernel<384, 128, 1, 2><<<(NN_SEDG + 31) / 32, 256, 0, stream>>>(
      NN_SEDG, out_super, sg0, out_super, sg1, superedges, nullptr,
      seW0, seW1, seW2, nullptr,
      Pse.b0, Pse.b1, Pse.b2, Pse.g0, Pse.be0, Pse.g1, Pse.be1, Pse.g2, Pse.be2,
      superedges, nullptr, out_sedges);

  // ---- 10) edge MLP (uses NEW nodes, +residual, tanh out) -> out_edges ----
  mlp_kernel<384, 128, 1, 2><<<(NN_EDGES + 31) / 32, 256, 0, stream>>>(
      NN_EDGES, out_nodes, g0i, out_nodes, g1i, edges, nullptr,
      eW0, eW1, eW2, nullptr,
      Pe.b0, Pe.b1, Pe.b2, Pe.g0, Pe.be0, Pe.g1, Pe.be1, Pe.g2, Pe.be2,
      edges, nullptr, out_edges);
}
